// DeepGemmFp8BlockLinear_5437428597395
// MI455X (gfx1250) — 8-bit matrix operand order repaired and checked by reading, compiled, not run on hardware
//
#include <hip/hip_runtime.h>
#include <hip/hip_bf16.h>


typedef __attribute__((ext_vector_type(16))) int   v16i;
typedef __attribute__((ext_vector_type(8)))  float v8f;

#define BLK   128
#define FP8MX 448.0f

__device__ __forceinline__ unsigned int f32_to_e4m3(float x) {
    unsigned int u  = __float_as_uint(x);
    unsigned int s  = (u >> 24) & 0x80u;
    unsigned int af = u & 0x7FFFFFFFu;
    if (af >= 0x7F800000u) return s | 0x7Fu;
    if (af < 0x3C800000u) {
        float q = rintf(__uint_as_float(af) * 512.0f);
        return s | (unsigned int)q;
    }
    unsigned int lsb = (af >> 20) & 1u;
    unsigned int r   = af + 0x0007FFFFu + lsb;
    int          e8  = (int)(r >> 23) - 120;
    unsigned int m8  = (r >> 20) & 7u;
    if (e8 > 15 || (e8 == 15 && m8 == 7)) return s | 0x7Eu;
    return s | (unsigned int)(e8 << 3) | m8;
}

__device__ __forceinline__ unsigned int pack4_e4m3(float a, float b, float c, float d) {
    return f32_to_e4m3(a) | (f32_to_e4m3(b) << 8) |
           (f32_to_e4m3(c) << 16) | (f32_to_e4m3(d) << 24);
}

__device__ __forceinline__ float bf16u_to_f32(unsigned short h) {
    return __uint_as_float(((unsigned int)h) << 16);
}

__global__ __launch_bounds__(256)
void quant_act_kernel(const __hip_bfloat16* __restrict__ X,
                      unsigned char* __restrict__ Aswz,
                      float* __restrict__ AsT,
                      int M, int K) {
    const int KB   = K / BLK;
    const int gid  = blockIdx.x * 8 + (threadIdx.x >> 5);
    const int lane = threadIdx.x & 31;
    const int m    = gid / KB;
    const int g    = gid - m * KB;
    if (m >= M) return;

    const size_t src = (size_t)m * K + (size_t)g * BLK + (size_t)lane * 4;
    ushort4 hh = *(const ushort4*)((const unsigned short*)X + src);
    float x0 = bf16u_to_f32(hh.x), x1 = bf16u_to_f32(hh.y);
    float x2 = bf16u_to_f32(hh.z), x3 = bf16u_to_f32(hh.w);

    float amax = fmaxf(fmaxf(fabsf(x0), fabsf(x1)), fmaxf(fabsf(x2), fabsf(x3)));
    #pragma unroll
    for (int off = 16; off > 0; off >>= 1)
        amax = fmaxf(amax, __shfl_xor(amax, off, 32));

    const float scale = fmaxf(amax, 1e-12f) * (1.0f / FP8MX);
    const float inv   = 1.0f / scale;
    const unsigned int packed = pack4_e4m3(x0 * inv, x1 * inv, x2 * inv, x3 * inv);

    const int kp   = lane * 4;
    const int g64  = kp >> 6;
    const int r    = kp & 63;
    const int p    = r >> 4;
    const int h2   = (r >> 3) & 1;
    const int v    = g64 * 8 + p * 2 + (lane & 1);
    const int c    = v >> 2;
    const int ldst = (m & 15) + 16 * h2;

    const size_t frag = ((size_t)(m >> 4) * KB + g) * 2048;
    *(unsigned int*)(Aswz + frag + c * 512 + ldst * 16 + (v & 3) * 4) = packed;

    if (lane == 0) AsT[(size_t)g * M + m] = scale;
}

__global__ __launch_bounds__(256)
void pack_weight_kernel(const float* __restrict__ Wf,
                        unsigned char* __restrict__ Bswz,
                        int N, int K) {
    const int KB = K / BLK;
    long long i = (long long)blockIdx.x * blockDim.x + threadIdx.x;
    long long total4 = ((long long)N * K) >> 2;
    if (i >= total4) return;

    float4 w = ((const float4*)Wf)[i];
    const unsigned int packed = pack4_e4m3(w.x, w.y, w.z, w.w);

    const long long idx = i << 2;
    const int n  = (int)(idx / K);
    const int k  = (int)(idx - (long long)n * K);
    const int kb = k >> 7;
    const int kp = k & 127;
    const int gg  = ((kp >> 6) << 1) | ((kp >> 5) & 1);
    const int rem = kp & 31;
    const int h2  = (rem >> 3) & 1;
    const int j   = (((rem >> 4) & 1) << 1) | ((rem >> 2) & 1);
    const int v   = 4 * gg + j;
    const int ldst = (n & 15) + 16 * h2;

    const size_t frag = ((size_t)(n >> 4) * KB + kb) * 2048;
    *(unsigned int*)(Bswz + frag + gg * 512 + ldst * 16 + (v & 3) * 4) = packed;
}

__global__ __launch_bounds__(256)
void fp8_block_gemm_kernel(const unsigned char* __restrict__ Aswz,
                           const float*         __restrict__ AsT,
                           const unsigned char* __restrict__ Bswz,
                           const float*         __restrict__ Ws,
                           __hip_bfloat16*      __restrict__ Out,
                           int M, int N, int K) {
    const int KB   = K / BLK;
    const int tid  = threadIdx.x;
    const int wave = tid >> 5;
    const int lane = tid & 31;
    const int half = lane >> 4;
    const int ln   = lane & 15;

    const int wm    = wave & 1;
    const int wn    = wave >> 1;
    const int Mbase = blockIdx.y * 128 + wm * 64;
    const int Nbase = blockIdx.x * 128 + wn * 32;
    const int nb    = blockIdx.x;

    const int mtile0 = (blockIdx.y * 8) + wm * 4;
    const int ntile0 = (blockIdx.x * 8) + wn * 2;

    v8f acc[4][2];
    #pragma unroll
    for (int mi = 0; mi < 4; ++mi)
        #pragma unroll
        for (int ni = 0; ni < 2; ++ni)
            acc[mi][ni] = (v8f){0.f, 0.f, 0.f, 0.f, 0.f, 0.f, 0.f, 0.f};

    for (int kb = 0; kb < KB; ++kb) {
        v16i a[4];
        #pragma unroll
        for (int mi = 0; mi < 4; ++mi) {
            const unsigned char* ap =
                Aswz + ((size_t)(mtile0 + mi) * KB + kb) * 2048 + (size_t)lane * 16;
            #pragma unroll
            for (int c = 0; c < 4; ++c) {
                const int4 t = *(const int4*)(ap + c * 512);
                a[mi][4 * c + 0] = t.x;
                a[mi][4 * c + 1] = t.y;
                a[mi][4 * c + 2] = t.z;
                a[mi][4 * c + 3] = t.w;
            }
        }

        v16i b[2];
        #pragma unroll
        for (int ni = 0; ni < 2; ++ni) {
            const unsigned char* bp =
                Bswz + ((size_t)(ntile0 + ni) * KB + kb) * 2048 + (size_t)lane * 16;
            #pragma unroll
            for (int c = 0; c < 4; ++c) {
                const int4 t = *(const int4*)(bp + c * 512);
                b[ni][4 * c + 0] = t.x;
                b[ni][4 * c + 1] = t.y;
                b[ni][4 * c + 2] = t.z;
                b[ni][4 * c + 3] = t.w;
            }
        }

        if (kb + 1 < KB) {
            __builtin_prefetch(Aswz + ((size_t)mtile0 * KB + kb + 1) * 2048 +
                               (size_t)lane * 16, 0, 1);
            __builtin_prefetch(Bswz + ((size_t)ntile0 * KB + kb + 1) * 2048 +
                               (size_t)lane * 16, 0, 1);
        }

        const float ws = Ws[(size_t)nb * KB + kb];

        #pragma unroll
        for (int mi = 0; mi < 4; ++mi) {
            const float* sp = AsT + (size_t)kb * M + (Mbase + mi * 16 + half * 8);
            const float4 sA = *(const float4*)sp;
            const float4 sB = *(const float4*)(sp + 4);
            float s8[8] = {sA.x * ws, sA.y * ws, sA.z * ws, sA.w * ws,
                           sB.x * ws, sB.y * ws, sB.z * ws, sB.w * ws};

            #pragma unroll
            for (int ni = 0; ni < 2; ++ni) {
                v8f z = {};
                v8f part = __builtin_amdgcn_wmma_f32_16x16x128_fp8_fp8(
                    a[mi], b[ni], (short)0, z, false, false);
                #pragma unroll
                for (int i = 0; i < 8; ++i)
                    acc[mi][ni][i] = fmaf(part[i], s8[i], acc[mi][ni][i]);
            }
        }
    }

    #pragma unroll
    for (int mi = 0; mi < 4; ++mi)
        #pragma unroll
        for (int ni = 0; ni < 2; ++ni)
            #pragma unroll
            for (int i = 0; i < 8; ++i) {
                const int m = Mbase + mi * 16 + i + half * 8;
                const int n = Nbase + ni * 16 + ln;
                Out[(size_t)m * N + n] = __float2bfloat16(acc[mi][ni][i]);
            }
}

extern "C" void kernel_launch(void* const* d_in, const int* in_sizes, int n_in,
                              void* d_out, int out_size, void* d_ws, size_t ws_size,
                              hipStream_t stream) {
    const __hip_bfloat16* inp     = (const __hip_bfloat16*)d_in[0];
    const float*          w_f32   = (const float*)d_in[1];
    const float*          w_scale = (const float*)d_in[2];

    const int K  = 7168;
    const int M  = in_sizes[0] / K;
    const int N  = in_sizes[1] / K;
    const int KB = K / BLK;

    unsigned char* Bswz = (unsigned char*)d_ws;
    unsigned char* Aswz = Bswz + (size_t)N * K;
    float*         AsT  = (float*)(Aswz + (size_t)M * K);

    {
        long long total4 = ((long long)N * K) / 4;
        int blocks = (int)((total4 + 255) / 256);
        pack_weight_kernel<<<blocks, 256, 0, stream>>>(w_f32, Bswz, N, K);
    }
    {
        long long groups = (long long)M * KB;
        int blocks = (int)((groups + 7) / 8);
        quant_act_kernel<<<blocks, 256, 0, stream>>>(inp, Aswz, AsT, M, K);
    }
    {
        dim3 grid(N / 128, M / 128);
        fp8_block_gemm_kernel<<<grid, 256, 0, stream>>>(
            Aswz, AsT, Bswz, w_scale, (__hip_bfloat16*)d_out, M, N, K);
    }
}
